// MHA_6073083756856
// MI455X (gfx1250) — compile-verified, matrix operand order checked by reading, not run on hardware
//
#include <hip/hip_runtime.h>
#include <hip/hip_bf16.h>

// ---------------------------------------------------------------------------
// MHA forward for gfx1250 (MI455X): f16 WMMA GEMMs and flash attention, both
// fed by double-buffered async-to-LDS staging (ASYNCcnt), fp32 softmax.
// D_MODEL=1024, N_HEADS=16, D_K=64, B=4, S=2048.
// ---------------------------------------------------------------------------

#define D_MODEL 1024
#define N_HEADS 16
#define D_K     64
#define BATCH   4
#define SEQ     2048
#define ROWS    (BATCH * SEQ)          // 8192
#define QKV_N   (3 * D_MODEL)          // 3072

typedef _Float16 h8  __attribute__((ext_vector_type(8)));
typedef _Float16 h16 __attribute__((ext_vector_type(16)));
typedef float    f8  __attribute__((ext_vector_type(8)));

// ln(10000)/64
#define ROPE_LOG_SCALE 0.14391156533f

// ---------------------------------------------------------------------------
// Fragment helpers (wave32).  A-fragment and B-fragment share the same lane
// layout: lane L holds row (L&15); halves 0..7 = K[k0 + 8*(L>>4) .. +7],
// halves 8..15 = K[k0 + 16 + 8*(L>>4) .. +7].  For B we load rows of B^T
// (i.e. W stored as [N][K] row-major), which matches x @ W^T GEMMs exactly.
// ---------------------------------------------------------------------------
__device__ __forceinline__ h16 load_frag(const _Float16* base,
                                         int ld, int r0, int k0, int lane) {
    const int hf = lane >> 4;
    const int mr = lane & 15;
    const _Float16* p = base + (size_t)(r0 + mr) * ld + (k0 + 8 * hf);
    h8 lo = *(const h8*)(p);
    h8 hi = *(const h8*)(p + 16);
    h16 r;
#pragma unroll
    for (int i = 0; i < 8; ++i) { r[i] = lo[i]; r[i + 8] = hi[i]; }
    return r;
}

__device__ __forceinline__ f8 wmma_f16(h16 a, h16 b, f8 c) {
    return __builtin_amdgcn_wmma_f32_16x16x32_f16(false, a, false, b,
                                                  (short)0, c, false, false);
}

// Async 16B copy global -> LDS (ASYNCcnt-tracked, CDNA5 §08).
__device__ __forceinline__ void async_cp16(unsigned lds, unsigned long long g) {
    asm volatile("global_load_async_to_lds_b128 %0, %1, off"
                 :: "v"(lds), "v"(g) : "memory");
}
__device__ __forceinline__ unsigned lds_addr(const void* p) {
    return (unsigned)(unsigned long long)p;
}

// reductions across the 16-lane group that holds one C-fragment row
__device__ __forceinline__ float redmax16(float x) {
#pragma unroll
    for (int m = 1; m < 16; m <<= 1) x = fmaxf(x, __shfl_xor(x, m, 32));
    return x;
}
__device__ __forceinline__ float redsum16(float x) {
#pragma unroll
    for (int m = 1; m < 16; m <<= 1) x += __shfl_xor(x, m, 32);
    return x;
}

// ---------------------------------------------------------------------------
// fp32 -> f16 conversion
// ---------------------------------------------------------------------------
__global__ void cvt_f32_to_f16(const float* __restrict__ src,
                               _Float16* __restrict__ dst, int n) {
    int i = blockIdx.x * 256 + threadIdx.x;
    if (i < n) dst[i] = (_Float16)src[i];
}

// ---------------------------------------------------------------------------
// LDS tiles for the staged GEMMs: block tile = 32 rows x 256 cols.
// Bt: 256 weight rows x 32 K-halves (padded to 40 => conflict-free b128 reads)
// At: 32 activation rows x 32 K-halves, duplicated (one copy per wave pair)
// Double-buffered along K.  ~50 KB/block.
// ---------------------------------------------------------------------------
struct __align__(16) GemmSmem {
    _Float16 Bt[2][256][40];
    _Float16 At[2][2][32][40];
};

// Pipelined K-loop: per K-step each wave issues 10 async b128 copies for the
// NEXT step; s_wait_asynccnt 10 guarantees the CURRENT batch has landed
// (async loads complete in order); barrier; 2 A-frags + 4 B-frags ds reads,
// 8 WMMAs; barrier.
__device__ __forceinline__ void gemm_k_loop(const _Float16* A,
                                            const _Float16* Bw,
                                            int rb, int cb0, GemmSmem& sm,
                                            int tid, int lane, int w,
                                            f8 acc[2][4]) {
    auto issue = [&](int k0, int buf) {
#pragma unroll
        for (int i = 0; i < 8; ++i) {          // 256x32 weight tile: 8/wave
            int id = i * 128 + tid;
            int r = id >> 2, c = id & 3;
            async_cp16(lds_addr(&sm.Bt[buf][r][c * 8]),
                       (unsigned long long)(Bw + (size_t)(cb0 + r) * D_MODEL +
                                            k0 + c * 8));
        }
#pragma unroll
        for (int i = 0; i < 2; ++i) {          // 32x32 A tile, 2 copies: 2/wave
            int id = i * 128 + tid;
            int cp = id >> 7, r = (id >> 2) & 31, c = id & 3;
            async_cp16(lds_addr(&sm.At[buf][cp][r][c * 8]),
                       (unsigned long long)(A + (size_t)(rb + r) * D_MODEL +
                                            k0 + c * 8));
        }
    };

    issue(0, 0);
    const int NK = D_MODEL / 32;
    const int cp = w >> 1;
    for (int kt = 0; kt < NK; ++kt) {
        const int cur = kt & 1;
        if (kt + 1 < NK) {
            issue((kt + 1) * 32, cur ^ 1);
            asm volatile("s_wait_asynccnt 10" ::: "memory");
        } else {
            asm volatile("s_wait_asynccnt 0" ::: "memory");
        }
        __syncthreads();
        h16 a0 = load_frag(&sm.At[cur][cp][0][0], 40, 0, 0, lane);
        h16 a1 = load_frag(&sm.At[cur][cp][0][0], 40, 16, 0, lane);
#pragma unroll
        for (int j = 0; j < 4; ++j) {
            h16 b = load_frag(&sm.Bt[cur][0][0], 40, w * 64 + 16 * j, 0, lane);
            acc[0][j] = wmma_f16(a0, b, acc[0][j]);
            acc[1][j] = wmma_f16(a1, b, acc[1][j]);
        }
        __syncthreads();
    }
}

// ---------------------------------------------------------------------------
// QKV projection + bias + RoPE + scatter into Q/K ([b,h,s,d]) and V^T
// ([b,h,d,s]).  Block = 4 waves computing 32 rows x 256 cols.
// grid = (QKV_N/256, ROWS/32)
// ---------------------------------------------------------------------------
__global__ __launch_bounds__(128) void qkv_rope_kernel(
    const _Float16* __restrict__ xh,     // [ROWS][1024]
    const _Float16* __restrict__ wh,     // [3072][1024]
    const float*    __restrict__ bias,   // [3072]
    const int*      __restrict__ tp,     // [SEQ]
    _Float16* __restrict__ Qb,           // [B][H][S][64]
    _Float16* __restrict__ Kb,           // [B][H][S][64]
    _Float16* __restrict__ Vt)           // [B][H][64][S]
{
    __shared__ GemmSmem sm;
    const int tid  = threadIdx.x;
    const int lane = tid & 31;
    const int w    = tid >> 5;
    const int cb0  = blockIdx.x * 256;
    const int cb   = cb0 + w * 64;       // 64-aligned => one head, one of q/k/v
    const int rb   = blockIdx.y * 32;

    f8 acc[2][4] = {};
    gemm_k_loop(xh, wh, rb, cb0, sm, tid, lane, w, acc);

    const int hf = lane >> 4;
    const int mr = lane & 15;
    const int which = cb >> 10;          // 0=q 1=k 2=v (wave-uniform)
    const int hh    = (cb >> 6) & 15;    // head (wave-uniform)

#pragma unroll
    for (int rt = 0; rt < 2; ++rt) {
#pragma unroll
        for (int j = 0; j < 4; ++j) {
            const int n = cb + 16 * j + mr;
            const int d = n & 63;
            const float bj = bias[n];
#pragma unroll
            for (int v = 0; v < 8; ++v) {
                const int row = rb + rt * 16 + v + 8 * hf;  // global token row
                const int bi  = row >> 11;                  // batch
                const int s   = row & (SEQ - 1);            // seq position
                const float val = acc[rt][j][v] + bj;
                if (which == 2) {
                    // V: no RoPE, store transposed [b][h][d][s]
                    Vt[(((size_t)bi * N_HEADS + hh) * D_K + d) * SEQ + s] =
                        (_Float16)val;
                } else {
                    // RoPE: pair (even d, odd d) lives in adjacent lanes
                    const float pos = (float)tp[s];
                    const float xp  = __shfl_xor(val, 1, 32);
                    const float invf =
                        __expf(-(float)(d & ~1) * ROPE_LOG_SCALE);
                    const float ang = pos * invf;
                    float sn, cs;
                    __sincosf(ang, &sn, &cs);
                    const float r = (d & 1) ? (xp * sn + val * cs)
                                            : (val * cs - xp * sn);
                    const size_t idx =
                        (((size_t)bi * N_HEADS + hh) * SEQ + s) * D_K + d;
                    if (which == 0) Qb[idx] = (_Float16)(r * 0.125f);
                    else            Kb[idx] = (_Float16)r;
                }
            }
        }
    }
}

// ---------------------------------------------------------------------------
// Causal flash attention.  Block = 4 waves = 64 queries of one (b,h); the
// 32-key K tile and V^T tile are shared by all waves and double-buffered in
// LDS via async copies.  Per step/wave: 4 async b128 issues, 4 QK^T WMMAs,
// fp32 online softmax (xor-shuffle row reductions), P restaged C->A through
// LDS, 4 PV WMMAs.  grid = (S/64, H, B).
// ---------------------------------------------------------------------------
struct __align__(16) FlashSmem {
    _Float16 Kt[2][32][72];    // 32 keys x 64 d (padded)
    _Float16 Vtile[2][64][40]; // 64 d x 32 keys (padded)
    _Float16 Pb[4][16][40];    // per-wave P tile
};

__global__ __launch_bounds__(128) void flash_attn_kernel(
    const _Float16* __restrict__ Qb,
    const _Float16* __restrict__ Kb,
    const _Float16* __restrict__ Vt,
    _Float16* __restrict__ attn)          // [ROWS][1024]
{
    __shared__ FlashSmem sm;

    const int tid  = threadIdx.x;
    const int lane = tid & 31;
    const int w    = tid >> 5;
    const int b    = blockIdx.z;
    const int h    = blockIdx.y;
    const int q0b  = blockIdx.x * 64;
    const int q0   = q0b + w * 16;
    const int hf   = lane >> 4;
    const int mr   = lane & 15;
    // wave-uniform copy for scalar branching (keeps EXEC all-ones at WMMAs)
    const int q0u  = __builtin_amdgcn_readfirstlane(q0);

    const _Float16* Q  = Qb + ((size_t)(b * N_HEADS + h)) * SEQ * D_K;
    const _Float16* Km = Kb + ((size_t)(b * N_HEADS + h)) * SEQ * D_K;
    const _Float16* V  = Vt + ((size_t)(b * N_HEADS + h)) * D_K * SEQ;

    const h16 qa0 = load_frag(Q, D_K, q0, 0, lane);
    const h16 qa1 = load_frag(Q, D_K, q0, 32, lane);

    f8 acc[4] = {};
    float m[8], l[8];
#pragma unroll
    for (int v = 0; v < 8; ++v) { m[v] = -1e30f; l[v] = 0.f; }

    auto issue = [&](int kb, int buf) {
#pragma unroll
        for (int i = 0; i < 2; ++i) {   // K tile: 32 rows x 128B
            int id = i * 128 + tid;
            int r = id >> 3, c = id & 7;
            async_cp16(lds_addr(&sm.Kt[buf][r][c * 8]),
                       (unsigned long long)(Km + (size_t)(kb + r) * D_K +
                                            c * 8));
        }
#pragma unroll
        for (int i = 0; i < 2; ++i) {   // V^T tile: 64 rows x 64B
            int id = i * 128 + tid;
            int r = id >> 2, c = id & 3;
            async_cp16(lds_addr(&sm.Vtile[buf][r][c * 8]),
                       (unsigned long long)(V + (size_t)r * SEQ + kb + c * 8));
        }
    };

    const int kbmax = q0b + 63;
    issue(0, 0);
    for (int kb = 0; kb <= kbmax; kb += 32) {
        const int cur = (kb >> 5) & 1;
        if (kb + 32 <= kbmax) {
            issue(kb + 32, cur ^ 1);
            asm volatile("s_wait_asynccnt 4" ::: "memory");
        } else {
            asm volatile("s_wait_asynccnt 0" ::: "memory");
        }
        __syncthreads();
        if (kb <= q0u + 15) {           // scalar, wave-uniform causal guard
            // ---- scores: 2 tiles x (K=64 via 2 chained WMMAs) ----
            f8 s0 = {}, s1 = {};
            h16 kf = load_frag(&sm.Kt[cur][0][0], 72, 0, 0, lane);
            s0 = wmma_f16(qa0, kf, s0);
            kf = load_frag(&sm.Kt[cur][0][0], 72, 0, 32, lane);
            s0 = wmma_f16(qa1, kf, s0);
            kf = load_frag(&sm.Kt[cur][0][0], 72, 16, 0, lane);
            s1 = wmma_f16(qa0, kf, s1);
            kf = load_frag(&sm.Kt[cur][0][0], 72, 16, 32, lane);
            s1 = wmma_f16(qa1, kf, s1);
            // ---- online softmax update (fp32), write P tile to LDS ----
#pragma unroll
            for (int v = 0; v < 8; ++v) {
                const int row = q0 + v + 8 * hf;
                float a0 = (kb + mr      <= row) ? s0[v] : -1e30f;
                float a1 = (kb + 16 + mr <= row) ? s1[v] : -1e30f;
                float bm = redmax16(fmaxf(a0, a1));
                float mn = fmaxf(m[v], bm);
                float scale = __expf(m[v] - mn);
                float p0 = __expf(a0 - mn);
                float p1 = __expf(a1 - mn);
                float rs = redsum16(p0 + p1);
                m[v] = mn;
                l[v] = l[v] * scale + rs;
                acc[0][v] *= scale; acc[1][v] *= scale;
                acc[2][v] *= scale; acc[3][v] *= scale;
                const int r = v + 8 * hf;
                sm.Pb[w][r][mr]      = (_Float16)p0;
                sm.Pb[w][r][16 + mr] = (_Float16)p1;
            }
            // ---- P (A-frag from LDS) x V^T (B-frag from LDS) ----
            const h16 pa = load_frag(&sm.Pb[w][0][0], 40, 0, 0, lane);
#pragma unroll
            for (int j = 0; j < 4; ++j) {
                h16 vf = load_frag(&sm.Vtile[cur][0][0], 40, 16 * j, 0, lane);
                acc[j] = wmma_f16(pa, vf, acc[j]);
            }
        }
        __syncthreads();
    }

    // ---- normalize and store to [b][s][h*64 + d] (f16) ----
#pragma unroll
    for (int v = 0; v < 8; ++v) {
        const float inv = 1.f / l[v];
        const int s = q0 + v + 8 * hf;
        const size_t base =
            ((size_t)b * SEQ + s) * D_MODEL + h * D_K + mr;
#pragma unroll
        for (int j = 0; j < 4; ++j)
            attn[base + 16 * j] = (_Float16)(acc[j][v] * inv);
    }
}

// ---------------------------------------------------------------------------
// Output projection: out = attn @ W_o^T + b_o  (fp32 result)
// grid = (D_MODEL/256, ROWS/32), 4 waves/block, LDS-staged like QKV.
// ---------------------------------------------------------------------------
__global__ __launch_bounds__(128) void oproj_kernel(
    const _Float16* __restrict__ ah,     // [ROWS][1024]
    const _Float16* __restrict__ wh,     // [1024][1024]
    const float*    __restrict__ bias,   // [1024]
    float* __restrict__ out)             // [ROWS][1024]
{
    __shared__ GemmSmem sm;
    const int tid  = threadIdx.x;
    const int lane = tid & 31;
    const int w    = tid >> 5;
    const int cb0  = blockIdx.x * 256;
    const int cb   = cb0 + w * 64;
    const int rb   = blockIdx.y * 32;

    f8 acc[2][4] = {};
    gemm_k_loop(ah, wh, rb, cb0, sm, tid, lane, w, acc);

    const int hf = lane >> 4;
    const int mr = lane & 15;
#pragma unroll
    for (int rt = 0; rt < 2; ++rt) {
#pragma unroll
        for (int j = 0; j < 4; ++j) {
            const int n = cb + 16 * j + mr;
            const float bj = bias[n];
#pragma unroll
            for (int v = 0; v < 8; ++v) {
                const int row = rb + rt * 16 + v + 8 * hf;
                out[(size_t)row * D_MODEL + n] = acc[rt][j][v] + bj;
            }
        }
    }
}

// ---------------------------------------------------------------------------
// Launch
// ---------------------------------------------------------------------------
extern "C" void kernel_launch(void* const* d_in, const int* in_sizes, int n_in,
                              void* d_out, int out_size, void* d_ws, size_t ws_size,
                              hipStream_t stream) {
    const float* x     = (const float*)d_in[0];
    const int*   tp    = (const int*)  d_in[1];
    const float* W_qkv = (const float*)d_in[2];
    const float* b_qkv = (const float*)d_in[3];
    const float* W_o   = (const float*)d_in[4];
    const float* b_o   = (const float*)d_in[5];
    float* out = (float*)d_out;

    // workspace carve-up (all 256B aligned)
    char* p = (char*)d_ws;
    auto take = [&](size_t bytes) {
        char* r = p;
        p += (bytes + 255) & ~(size_t)255;
        return r;
    };
    _Float16* xh    = (_Float16*)take((size_t)ROWS  * D_MODEL * 2); // 16 MB
    _Float16* wqkvh = (_Float16*)take((size_t)QKV_N * D_MODEL * 2); //  6 MB
    _Float16* woh   = (_Float16*)take((size_t)D_MODEL * D_MODEL * 2);
    _Float16* Qb    = (_Float16*)take((size_t)BATCH * N_HEADS * SEQ * D_K * 2);
    _Float16* Kb    = (_Float16*)take((size_t)BATCH * N_HEADS * SEQ * D_K * 2);
    _Float16* Vt    = (_Float16*)take((size_t)BATCH * N_HEADS * D_K * SEQ * 2);
    _Float16* attn  = (_Float16*)take((size_t)ROWS * D_MODEL * 2);

    {
        int n = ROWS * D_MODEL;
        cvt_f32_to_f16<<<(n + 255) / 256, 256, 0, stream>>>(x, xh, n);
    }
    {
        int n = QKV_N * D_MODEL;
        cvt_f32_to_f16<<<(n + 255) / 256, 256, 0, stream>>>(W_qkv, wqkvh, n);
    }
    {
        int n = D_MODEL * D_MODEL;
        cvt_f32_to_f16<<<(n + 255) / 256, 256, 0, stream>>>(W_o, woh, n);
    }

    qkv_rope_kernel<<<dim3(QKV_N / 256, ROWS / 32), 128, 0, stream>>>(
        xh, wqkvh, b_qkv, tp, Qb, Kb, Vt);

    flash_attn_kernel<<<dim3(SEQ / 64, N_HEADS, BATCH), 128, 0, stream>>>(
        Qb, Kb, Vt, attn);

    oproj_kernel<<<dim3(D_MODEL / 256, ROWS / 32), 128, 0, stream>>>(
        attn, woh, b_o, out);
}
